// OnlineKNN_13795434954899
// MI455X (gfx1250) — matrix operand order repaired and checked by reading, not compiled, not run on hardware
//
#include <hip/hip_runtime.h>
#include <hip/hip_bf16.h>
#include <math.h>

typedef __attribute__((ext_vector_type(16))) __bf16 v16bf;
typedef __attribute__((ext_vector_type(8)))  __bf16 v8bf;
typedef __attribute__((ext_vector_type(8)))  float  v8f;

#define N_ROWS 1024
#define DIM    512
#define KQ     65536
#define STRIPE 256
#define KNN    200
#define NCLS   1000
#define NBINS  4096
#define CAND   2048

// ---------------- split f32 -> bf16 hi/lo ----------------
__global__ void cvt_split(const float* __restrict__ src,
                          __bf16* __restrict__ hi, __bf16* __restrict__ lo, int n) {
  int i = blockIdx.x * blockDim.x + threadIdx.x;
  if (i < n) {
    float f = src[i];
    __bf16 h = (__bf16)f;
    hi[i] = h;
    lo[i] = (__bf16)(f - (float)h);
  }
}

// ---------------- sim = F · Q^T  (split-bf16, 3 WMMAs per tile-step) ----
// wave computes a 16(M) x 64(N) tile; 8 waves/block; grid exact (EXEC all-1s)
__global__ void __launch_bounds__(256) gemm_stripe(
    const __bf16* __restrict__ featH, const __bf16* __restrict__ featL,
    const __bf16* __restrict__ qH,    const __bf16* __restrict__ qL,
    float* __restrict__ sim, int mBase)
{
  int wave = (blockIdx.x * 256 + threadIdx.x) >> 5;
  int lane = threadIdx.x & 31;
  int h    = lane >> 4;     // lane half
  int lm   = lane & 15;
  int mt   = wave >> 10;            // 0..15  (STRIPE/16 M-tiles)
  int nb   = (wave & 1023) << 6;    // N base, 1024 wave-columns * 64

  const __bf16* paH = featH + (size_t)(mBase + mt * 16 + lm) * DIM;
  const __bf16* paL = featL + (size_t)(mBase + mt * 16 + lm) * DIM;
  const __bf16* pbH = qH + (size_t)(nb + lm) * DIM;
  const __bf16* pbL = qL + (size_t)(nb + lm) * DIM;

  v8f acc[4] = {v8f{}, v8f{}, v8f{}, v8f{}};

  for (int k0 = 0; k0 < DIM; k0 += 32) {
    union { v16bf v; v8bf p[2]; } ah, al;
    // A 16x32 bf16 layout: half h -> K in {16h..16h+15} contiguous
    ah.v = *(const v16bf*)(paH + k0 + 16 * h);
    al.v = *(const v16bf*)(paL + k0 + 16 * h);
#pragma unroll
    for (int sub = 0; sub < 4; ++sub) {
      // B 32x16 bf16 layout: half h -> K in {16h..16h+15} contiguous, N = lm
      v16bf bh = *(const v16bf*)(pbH + (size_t)sub * 16 * DIM + k0 + 16 * h);
      v16bf bl = *(const v16bf*)(pbL + (size_t)sub * 16 * DIM + k0 + 16 * h);
      acc[sub] = __builtin_amdgcn_wmma_f32_16x16x32_bf16(false, ah.v, false, bh,
                                                         (short)0, acc[sub], false, false);
      acc[sub] = __builtin_amdgcn_wmma_f32_16x16x32_bf16(false, ah.v, false, bl,
                                                         (short)0, acc[sub], false, false);
      acc[sub] = __builtin_amdgcn_wmma_f32_16x16x32_bf16(false, al.v, false, bh,
                                                         (short)0, acc[sub], false, false);
    }
  }

  // D layout: lane -> N = lm ; VGPR r -> M = r + 8h
  int mLoc = mt * 16 + 8 * h;
#pragma unroll
  for (int sub = 0; sub < 4; ++sub) {
    int n = nb + sub * 16 + lm;
#pragma unroll
    for (int r = 0; r < 8; ++r)
      sim[(size_t)(mLoc + r) * KQ + n] = acc[sub][r];
  }
}

// ---------------- per-row top-200 + class voting + argmax ----------------
__global__ void __launch_bounds__(256) topk_stripe(
    const float* __restrict__ sim, const int* __restrict__ labels,
    const int* __restrict__ qlabels, int mBase, unsigned* __restrict__ hits)
{
  __shared__ unsigned hist[NBINS];
  __shared__ float    cv[CAND];
  __shared__ int      ci[CAND];
  __shared__ float    scores[NCLS];
  __shared__ float    rv[256];
  __shared__ int      ri[256];
  __shared__ unsigned cnt;
  __shared__ int      tbin;

  int tid  = threadIdx.x;
  int lrow = blockIdx.x;
  const float* row = sim + (size_t)lrow * KQ;

  for (int b = tid; b < NBINS; b += 256) hist[b] = 0u;
  if (tid == 0) cnt = 0u;
  __syncthreads();

  // pass 1: histogram of sims in [-1,1]
  for (int i = tid; i < KQ; i += 256) {
    float v = row[i];
    int b = (int)((v + 1.0f) * 2048.0f);
    b = b < 0 ? 0 : (b > NBINS - 1 ? NBINS - 1 : b);
    atomicAdd(&hist[b], 1u);
  }
  __syncthreads();

  if (tid == 0) {
    unsigned cum = 0; int b = NBINS - 1;
    for (; b > 0; --b) { cum += hist[b]; if (cum >= KNN) break; }
    tbin = b;
  }
  __syncthreads();
  int tb = tbin;

  // pass 2: collect candidates with bin >= threshold bin
  for (int i = tid; i < KQ; i += 256) {
    float v = row[i];
    int b = (int)((v + 1.0f) * 2048.0f);
    b = b < 0 ? 0 : (b > NBINS - 1 ? NBINS - 1 : b);
    if (b >= tb) {
      unsigned p = atomicAdd(&cnt, 1u);
      if (p < CAND) { cv[p] = v; ci[p] = i; }
    }
  }
  __syncthreads();
  unsigned C = cnt; if (C > CAND) C = CAND;
  for (unsigned p = C + tid; p < CAND; p += 256) { cv[p] = -3.0e38f; ci[p] = 0; }
  __syncthreads();

  // bitonic sort, descending by value
  for (int k = 2; k <= CAND; k <<= 1) {
    for (int j = k >> 1; j > 0; j >>= 1) {
      for (int i = tid; i < CAND; i += 256) {
        int l = i ^ j;
        if (l > i) {
          bool up = ((i & k) == 0);
          float a = cv[i], b2 = cv[l];
          if ((a < b2) == up) {
            cv[i] = b2; cv[l] = a;
            int t2 = ci[i]; ci[i] = ci[l]; ci[l] = t2;
          }
        }
      }
      __syncthreads();
    }
  }

  // exp-weighted class voting over exact top-200
  for (int c = tid; c < NCLS; c += 256) scores[c] = 0.0f;
  __syncthreads();
  if (tid < KNN) {
    float w = expf(cv[tid] / 0.07f);
    int cls = qlabels[ci[tid]];
    atomicAdd(&scores[cls], w);
  }
  __syncthreads();

  // argmax with first-max (smallest index) tie-break
  float bv = -1.0f; int bi = 0;
  for (int c = tid; c < NCLS; c += 256) {
    float v = scores[c];
    if (v > bv) { bv = v; bi = c; }
  }
  rv[tid] = bv; ri[tid] = bi;
  __syncthreads();
  for (int s = 128; s > 0; s >>= 1) {
    if (tid < s) {
      float ov = rv[tid + s]; int oi = ri[tid + s];
      if (ov > rv[tid] || (ov == rv[tid] && oi < ri[tid])) { rv[tid] = ov; ri[tid] = oi; }
    }
    __syncthreads();
  }
  if (tid == 0) {
    if (ri[0] == labels[mBase + lrow]) atomicAdd(hits, 1u);
  }
}

// ---------------- queue scatter-update into d_out ----------------
__global__ void copy_qf(const float* __restrict__ feat, const float* __restrict__ qf,
                        const int* __restrict__ pptr, float* __restrict__ out)
{
  size_t i = (size_t)blockIdx.x * blockDim.x + threadIdx.x; // < KQ*DIM
  int ptr = *pptr;
  int r = (int)(i >> 9);
  int c = (int)(i & 511);
  int d = r - ptr;
  out[i] = (d >= 0 && d < N_ROWS) ? feat[d * DIM + c] : qf[i];
}

__global__ void copy_ql(const int* __restrict__ labels, const int* __restrict__ qlab,
                        const int* __restrict__ pptr, float* __restrict__ out)
{
  int i = blockIdx.x * blockDim.x + threadIdx.x;
  int ptr = *pptr;
  int d = i - ptr;
  out[i] = (float)((d >= 0 && d < N_ROWS) ? labels[d] : qlab[i]);
}

__global__ void init_hits(unsigned* hits) { if (threadIdx.x == 0) *hits = 0u; }

__global__ void finalize(const unsigned* __restrict__ hits, const int* __restrict__ pptr,
                         float* __restrict__ acc_out, float* __restrict__ ptr_out)
{
  *acc_out = (float)(*hits) * (1.0f / (float)N_ROWS);
  *ptr_out = (float)(((*pptr) + N_ROWS) % KQ);
}

// ---------------- launch ----------------
extern "C" void kernel_launch(void* const* d_in, const int* in_sizes, int n_in,
                              void* d_out, int out_size, void* d_ws, size_t ws_size,
                              hipStream_t stream) {
  (void)in_sizes; (void)n_in; (void)out_size; (void)ws_size;
  const float* feat = (const float*)d_in[0];
  const float* qf   = (const float*)d_in[1];
  const int*   lab  = (const int*)d_in[2];
  const int*   qlab = (const int*)d_in[3];
  const int*   pptr = (const int*)d_in[4];

  float* out     = (float*)d_out;
  float* out_acc = out;
  float* out_qf  = out + 1;
  float* out_ql  = out + 1 + (size_t)KQ * DIM;
  float* out_ptr = out + 1 + (size_t)KQ * DIM + KQ;

  char* ws = (char*)d_ws;
  float*  sim   = (float*)ws;                                   // 64 MB stripe
  __bf16* featH = (__bf16*)(ws + (size_t)STRIPE * KQ * 4);
  __bf16* featL = featH + (size_t)N_ROWS * DIM;
  __bf16* qH    = featL + (size_t)N_ROWS * DIM;
  __bf16* qL    = qH + (size_t)KQ * DIM;
  unsigned* hits = (unsigned*)(qL + (size_t)KQ * DIM);

  init_hits<<<1, 32, 0, stream>>>(hits);
  cvt_split<<<(N_ROWS * DIM) / 256, 256, 0, stream>>>(feat, featH, featL, N_ROWS * DIM);
  cvt_split<<<(KQ * DIM) / 256, 256, 0, stream>>>(qf, qH, qL, KQ * DIM);
  copy_qf<<<(KQ * DIM) / 256, 256, 0, stream>>>(feat, qf, pptr, out_qf);
  copy_ql<<<KQ / 256, 256, 0, stream>>>(lab, qlab, pptr, out_ql);

  for (int s = 0; s < N_ROWS / STRIPE; ++s) {
    int mBase = s * STRIPE;
    gemm_stripe<<<(STRIPE / 16) * (KQ / 64) / 8, 256, 0, stream>>>(featH, featL, qH, qL, sim, mBase);
    topk_stripe<<<STRIPE, 256, 0, stream>>>(sim, lab, qlab, mBase, hits);
  }
  finalize<<<1, 1, 0, stream>>>(hits, pptr, out_acc, out_ptr);
}
